// _DeformConv_86260123172975
// MI455X (gfx1250) — matrix operand order repaired and checked by reading, not compiled, not run on hardware
//
#include <hip/hip_runtime.h>

#define CDNA_H  80
#define CDNA_W  80
#define CDNA_HW 6400
#define CDNA_CI 256
#define CDNA_CO 256
#define CDNA_B  4
#define NPIX    25600

#define AS 40   // LDS row stride (halfs) for A tiles: 80 B, 16B-aligned, bank-rotating
#define BS 40   // LDS row stride (halfs) for B tiles

typedef __attribute__((ext_vector_type(16))) _Float16 v16h;
typedef __attribute__((ext_vector_type(8)))  _Float16 v8h;
typedef __attribute__((ext_vector_type(8)))  float    v8f;

union HFrag { v8h h[2]; v16h v; };

struct MetaP { unsigned short s[4]; float w[4]; };   // 24 B

// ---- CDNA5 async helpers ---------------------------------------------------
__device__ __forceinline__ void async_ld_b128(unsigned lds_off, const void* g) {
  asm volatile("global_load_async_to_lds_b128 %0, %1, off"
               :: "v"(lds_off), "v"((unsigned long long)(uintptr_t)g)
               : "memory");
}
__device__ __forceinline__ void async_ld_b64(unsigned lds_off, const void* g) {
  asm volatile("global_load_async_to_lds_b64 %0, %1, off"
               :: "v"(lds_off), "v"((unsigned long long)(uintptr_t)g)
               : "memory");
}
__device__ __forceinline__ void wait_async0() {
#if __has_builtin(__builtin_amdgcn_s_wait_asynccnt)
  __builtin_amdgcn_s_wait_asynccnt(0);
#else
  asm volatile("s_wait_asynccnt 0x0" ::: "memory");
#endif
}
// LDS byte offset of a __shared__ object: flat LDS addresses carry the
// wave-relative LDS offset in their low 32 bits (ISA 10.2 aperture rules).
__device__ __forceinline__ unsigned lds_off(const void* p) {
  return (unsigned)(uintptr_t)p;
}

// ---------------------------------------------------------------------------
// Kernel 1: NCHW -> NHWC transpose of x (fp32), tiled through LDS.
// ---------------------------------------------------------------------------
__global__ __launch_bounds__(256) void k_transpose(const float* __restrict__ x,
                                                   float* __restrict__ xT) {
  __shared__ float tile[32][33];
  const int b   = blockIdx.z;
  const int hw0 = blockIdx.x * 32;
  const int c0  = blockIdx.y * 32;
  const int tx = threadIdx.x, ty = threadIdx.y;   // 32 x 8
#pragma unroll
  for (int i = 0; i < 4; ++i) {
    tile[ty + 8 * i][tx] =
        x[((size_t)(b * CDNA_CI + c0 + ty + 8 * i)) * CDNA_HW + hw0 + tx];
  }
  __syncthreads();
#pragma unroll
  for (int i = 0; i < 4; ++i) {
    xT[((size_t)(b * CDNA_HW + hw0 + ty + 8 * i)) * CDNA_CI + c0 + tx] =
        tile[tx][ty + 8 * i];
  }
}

// ---------------------------------------------------------------------------
// Kernel 2: weight prep.
//  Bw[kk][cb][o][c']  f16 = w_dcn[o][cb*32+c'][kk]          (main GEMM B)
//  Bo[kk*8+cb][o][c'] f16 = o<27 ? w_off[o][cb*32+c'][kk]:0 (offset GEMM B)
//  scale[o] = gamma/sqrt(var+eps);  bias2[o] = (b_dcn-mean)*scale + beta
// ---------------------------------------------------------------------------
__global__ __launch_bounds__(256) void k_prep(
    const float* __restrict__ w_dcn, const float* __restrict__ w_off,
    const float* __restrict__ b_dcn, const float* __restrict__ gamma,
    const float* __restrict__ beta,  const float* __restrict__ rmean,
    const float* __restrict__ rvar,
    _Float16* __restrict__ Bw, _Float16* __restrict__ Bo,
    float* __restrict__ scale, float* __restrict__ bias2) {
  const int id = blockIdx.x * 256 + threadIdx.x;
  if (id < 9 * 8 * 256 * 32) {
    const int cp = id & 31;
    const int o  = (id >> 5) & 255;
    const int cb = (id >> 13) & 7;
    const int kk = id >> 16;
    const int c  = cb * 32 + cp;
    Bw[id] = (_Float16)w_dcn[((size_t)o * CDNA_CI + c) * 9 + kk];
  }
  if (id < 72 * 1024) {
    const int step = id >> 10;
    const int rem  = id & 1023;
    const int o    = rem >> 5;
    const int cp   = rem & 31;
    const int kk = step >> 3, cb = step & 7;
    const int c  = cb * 32 + cp;
    Bo[id] = (o < 27) ? (_Float16)w_off[((size_t)o * CDNA_CI + c) * 9 + kk]
                      : (_Float16)0.f;
  }
  if (id < CDNA_CO) {
    const float inv = gamma[id] * rsqrtf(rvar[id] + 1e-5f);
    scale[id] = inv;
    bias2[id] = (b_dcn[id] - rmean[id]) * inv + beta[id];
  }
}

// ---------------------------------------------------------------------------
// Kernel 3: offset conv as WMMA GEMM (M=64 px/WG, N=32 (27 used), K=2304)
// then transform 27 channels -> py/px/sigmoid(mask) per (pixel, tap).
// ---------------------------------------------------------------------------
__global__ __launch_bounds__(256) void k_off_wmma(
    const float* __restrict__ xT, const _Float16* __restrict__ Bo,
    const float* __restrict__ b_off,
    float* __restrict__ pyA, float* __restrict__ pxA, float* __restrict__ mA) {
  __shared__ _Float16 sA1[64 * AS];    // 5.0 KB
  __shared__ _Float16 sB1[32 * BS];    // 2.5 KB
  __shared__ float    sOut[64 * 33];   // 8.25 KB

  const int t    = threadIdx.x;
  const int lane = t & 31;
  const int wv   = t >> 5;
  const int pix0 = blockIdx.x * 64;
  const int bidx = pix0 / CDNA_HW;
  const int bbase = bidx * CDNA_HW;

  const int arow = t & 63;
  const int aoct = t >> 6;
  const int prow = pix0 + arow;
  const int sprow = prow - bbase;
  const int hrow = sprow / CDNA_W;
  const int wrow = sprow - hrow * CDNA_W;

  const int msub = wv >> 1;
  const int nsub = wv & 1;

  v8f acc = {0.f, 0.f, 0.f, 0.f, 0.f, 0.f, 0.f, 0.f};

  for (int s = 0; s < 72; ++s) {
    const int kk = s >> 3, cb = s & 7;
    // ---- stage A: im2col row (zero-padded border), f32 -> f16 ----
    {
      const int yy = hrow - 1 + kk / 3;
      const int xx = wrow - 1 + kk % 3;
      float r[8] = {0.f, 0.f, 0.f, 0.f, 0.f, 0.f, 0.f, 0.f};
      if ((unsigned)yy < (unsigned)CDNA_H && (unsigned)xx < (unsigned)CDNA_W) {
        const float4* pp = (const float4*)(xT +
            ((size_t)(bbase + yy * CDNA_W + xx)) * CDNA_CI + cb * 32 + aoct * 8);
        const float4 a = pp[0], b = pp[1];
        r[0] = a.x; r[1] = a.y; r[2] = a.z; r[3] = a.w;
        r[4] = b.x; r[5] = b.y; r[6] = b.z; r[7] = b.w;
      }
      v8h hv;
#pragma unroll
      for (int i = 0; i < 8; ++i) hv[i] = (_Float16)r[i];
      *(v8h*)&sA1[arow * AS + aoct * 8] = hv;
    }
    // ---- stage B: 2 KB async copy into padded rows ----
    {
      const _Float16* src = Bo + (size_t)s * 1024;
      const int o = t >> 3, part = t & 7;          // 256 chunks of 4 halfs
      async_ld_b64(lds_off(&sB1[o * BS + part * 4]), src + t * 4);
    }
    wait_async0();
    __syncthreads();
    // ---- WMMA ----
    {
      const int m     = msub * 16 + (lane & 15);
      const int abase = (lane & 16) ? 16 : 0;
      HFrag af;
      af.h[0] = *(const v8h*)&sA1[m * AS + abase];
      af.h[1] = *(const v8h*)&sA1[m * AS + abase + 8];
      const int kb = (lane & 16) ? 16 : 0;
      const int o  = nsub * 16 + (lane & 15);
      HFrag bf;
      bf.v = *(const v16h*)&sB1[o * BS + kb];
      acc = __builtin_amdgcn_wmma_f32_16x16x32_f16(
          false, af.v, false, bf.v, (short)0, acc, false, false);
    }
    __syncthreads();
  }

  // ---- dump tiles to LDS, transform to sampling coordinates ----
  {
    const int o = nsub * 16 + (lane & 15);
    const float bo = (o < 27) ? b_off[o] : 0.f;
    const int m0 = msub * 16 + ((lane & 16) ? 8 : 0);
#pragma unroll
    for (int i = 0; i < 8; ++i) sOut[(m0 + i) * 33 + o] = acc[i] + bo;
  }
  __syncthreads();
  for (int e = t; e < 576; e += 256) {
    const int row = e / 9, k = e - row * 9;
    const int p  = pix0 + row;
    const int sp = p - bbase;
    const int h  = sp / CDNA_W;
    const int w  = sp - h * CDNA_W;
    const float dy = sOut[row * 33 + 2 * k];
    const float dx = sOut[row * 33 + 2 * k + 1];
    const float mm = 1.f / (1.f + __expf(-sOut[row * 33 + 18 + k]));
    pyA[p * 9 + k] = (float)(h - 1 + k / 3) + dy;
    pxA[p * 9 + k] = (float)(w - 1 + k % 3) + dx;
    mA[p * 9 + k]  = mm;
  }
}

// ---------------------------------------------------------------------------
// Kernel 4: fused bilinear-sample + GEMM (WMMA f16->f32) + BN + ReLU.
// 64 pixels x 256 outputs per WG; K = 9 taps x 256 ch; double-buffered LDS;
// B tiles fetched with global_load_async_to_lds_b128.
// ---------------------------------------------------------------------------
__global__ __launch_bounds__(256) void k_dcn_main(
    const float* __restrict__ xT, const _Float16* __restrict__ Bw,
    const float* __restrict__ pyA, const float* __restrict__ pxA,
    const float* __restrict__ mA,
    const float* __restrict__ scale, const float* __restrict__ bias2,
    float* __restrict__ out) {
  __shared__ MetaP    meta[576];          // 13.5 KB
  __shared__ _Float16 sA[2][64 * AS];     // 10 KB
  __shared__ _Float16 sB[2][256 * BS];    // 40 KB   (total 65,024 B)

  const int t    = threadIdx.x;
  const int lane = t & 31;
  const int wv   = t >> 5;
  const int pix0 = blockIdx.x * 64;
  const int bidx = pix0 / CDNA_HW;
  const int sp0  = pix0 - bidx * CDNA_HW;
  const int bbase = bidx * CDNA_HW;

  // Precompute bilinear corners; fold validity & sigmoid-mask into weights.
  for (int e = t; e < 576; e += 256) {
    const int row = e / 9, k = e - row * 9;
    const int pix = pix0 + row;
    const float py = pyA[pix * 9 + k];
    const float px = pxA[pix * 9 + k];
    const float m  = mA[pix * 9 + k];
    const float fy = floorf(py), fx = floorf(px);
    const int y0 = (int)fy, x0 = (int)fx;
    const float ly = py - fy, lx = px - fx;
    const float wt[4] = { (1.f - ly) * (1.f - lx) * m, (1.f - ly) * lx * m,
                          ly * (1.f - lx) * m,         ly * lx * m };
    MetaP mt;
#pragma unroll
    for (int j = 0; j < 4; ++j) {
      const int yy = y0 + (j >> 1), xx = x0 + (j & 1);
      const bool v = (yy >= 0 && yy < CDNA_H && xx >= 0 && xx < CDNA_W);
      const int cy = min(max(yy, 0), CDNA_H - 1);
      const int cx = min(max(xx, 0), CDNA_W - 1);
      mt.s[j] = (unsigned short)(cy * CDNA_W + cx);
      mt.w[j] = v ? wt[j] : 0.f;
    }
    meta[e] = mt;
  }
  __syncthreads();

  v8f acc[8];
  const v8f vzero = {0.f, 0.f, 0.f, 0.f, 0.f, 0.f, 0.f, 0.f};
#pragma unroll
  for (int j = 0; j < 8; ++j) acc[j] = vzero;

  const int msub  = wv >> 1;
  const int nhalf = wv & 1;
  const int arow  = t & 63;
  const int aoct  = t >> 6;

  // ---- staging helpers ----
  auto stageB = [&](int buf, int s) {
    const _Float16* src = Bw + (size_t)s * 8192;
#pragma unroll
    for (int i = 0; i < 4; ++i) {
      const int g = t + i * 256;                 // 1024 chunks of 8 halfs
      const int o = g >> 2, part = g & 3;
      async_ld_b128(lds_off(&sB[buf][o * BS + part * 8]), src + g * 8);
    }
  };
  auto stageA = [&](int buf, int s) {
    const int kk = s >> 3, cb = s & 7;
    const MetaP mt = meta[arow * 9 + kk];
    const int c0 = cb * 32 + aoct * 8;
    float4 va[4], vb[4];
#pragma unroll
    for (int j = 0; j < 4; ++j) {
      const float4* pp = (const float4*)(xT +
          ((size_t)(bbase + mt.s[j])) * CDNA_CI + c0);
      va[j] = pp[0];
      vb[j] = pp[1];
    }
    float r[8] = {0.f, 0.f, 0.f, 0.f, 0.f, 0.f, 0.f, 0.f};
#pragma unroll
    for (int j = 0; j < 4; ++j) {
      const float wj = mt.w[j];
      r[0] += wj * va[j].x; r[1] += wj * va[j].y;
      r[2] += wj * va[j].z; r[3] += wj * va[j].w;
      r[4] += wj * vb[j].x; r[5] += wj * vb[j].y;
      r[6] += wj * vb[j].z; r[7] += wj * vb[j].w;
    }
    v8h hv;
#pragma unroll
    for (int i = 0; i < 8; ++i) hv[i] = (_Float16)r[i];
    *(v8h*)&sA[buf][arow * AS + aoct * 8] = hv;
  };
  auto consume = [&](int buf) {
    const int m     = msub * 16 + (lane & 15);
    const int abase = (lane & 16) ? 16 : 0;
    HFrag af;
    af.h[0] = *(const v8h*)&sA[buf][m * AS + abase];
    af.h[1] = *(const v8h*)&sA[buf][m * AS + abase + 8];
    const int kb = (lane & 16) ? 16 : 0;
#pragma unroll
    for (int j = 0; j < 8; ++j) {
      const int o = nhalf * 128 + j * 16 + (lane & 15);
      HFrag bf;
      bf.v = *(const v16h*)&sB[buf][o * BS + kb];
      acc[j] = __builtin_amdgcn_wmma_f32_16x16x32_f16(
          false, af.v, false, bf.v, (short)0, acc[j], false, false);
    }
  };

  // ---- software-pipelined main loop (double-buffered, 1 barrier/step) ----
  stageB(0, 0);
  stageA(0, 0);
  for (int s = 0; s < 72; ++s) {
    const int buf = s & 1;
    wait_async0();          // own async B-tile for `buf` has landed
    __syncthreads();        // everyone's A stores landed; next buf fully read
    if (s + 1 < 72) {
      stageB(buf ^ 1, s + 1);
      stageA(buf ^ 1, s + 1);
    }
    consume(buf);
  }

  // ---- epilogue: BN + ReLU; per lane o is fixed and the 8 acc elements are
  // consecutive spatial positions -> two aligned b128 stores per tile ----
  const int spb = sp0 + msub * 16 + ((lane & 16) ? 8 : 0);
#pragma unroll
  for (int j = 0; j < 8; ++j) {
    const int o = nhalf * 128 + j * 16 + (lane & 15);
    const float s  = scale[o];
    const float bb = bias2[o];
    float* p = out + ((size_t)(bidx * CDNA_CO + o)) * CDNA_HW + spb;
    float4 r0, r1;
    r0.x = fmaxf(acc[j][0] * s + bb, 0.f);
    r0.y = fmaxf(acc[j][1] * s + bb, 0.f);
    r0.z = fmaxf(acc[j][2] * s + bb, 0.f);
    r0.w = fmaxf(acc[j][3] * s + bb, 0.f);
    r1.x = fmaxf(acc[j][4] * s + bb, 0.f);
    r1.y = fmaxf(acc[j][5] * s + bb, 0.f);
    r1.z = fmaxf(acc[j][6] * s + bb, 0.f);
    r1.w = fmaxf(acc[j][7] * s + bb, 0.f);
    *(float4*)p       = r0;
    *(float4*)(p + 4) = r1;
  }
}

// ---------------------------------------------------------------------------
extern "C" void kernel_launch(void* const* d_in, const int* in_sizes, int n_in,
                              void* d_out, int out_size, void* d_ws,
                              size_t ws_size, hipStream_t stream) {
  (void)in_sizes; (void)n_in; (void)out_size; (void)ws_size;
  const float* x     = (const float*)d_in[0];
  const float* w_off = (const float*)d_in[1];
  const float* b_off = (const float*)d_in[2];
  const float* w_dcn = (const float*)d_in[3];
  const float* b_dcn = (const float*)d_in[4];
  const float* gamma = (const float*)d_in[5];
  const float* beta  = (const float*)d_in[6];
  const float* rmean = (const float*)d_in[7];
  const float* rvar  = (const float*)d_in[8];
  float* out = (float*)d_out;

  char* ws = (char*)d_ws;
  size_t off = 0;
  auto take = [&](size_t bytes) -> void* {
    off = (off + 255) & ~(size_t)255;
    void* p = ws + off;
    off += bytes;
    return p;
  };
  float*    xT  = (float*)take((size_t)CDNA_B * CDNA_HW * CDNA_CI * 4);
  float*    pyA = (float*)take((size_t)NPIX * 9 * 4);
  float*    pxA = (float*)take((size_t)NPIX * 9 * 4);
  float*    mA  = (float*)take((size_t)NPIX * 9 * 4);
  _Float16* Bw  = (_Float16*)take((size_t)9 * 8 * 256 * 32 * 2);
  _Float16* Bo  = (_Float16*)take((size_t)72 * 1024 * 2);
  float*    scl = (float*)take(256 * 4);
  float*    bs2 = (float*)take(256 * 4);

  k_transpose<<<dim3(CDNA_HW / 32, CDNA_CI / 32, CDNA_B), dim3(32, 8), 0,
                stream>>>(x, xT);
  k_prep<<<(9 * 8 * 256 * 32 + 255) / 256, 256, 0, stream>>>(
      w_dcn, w_off, b_dcn, gamma, beta, rmean, rvar, Bw, Bo, scl, bs2);
  k_off_wmma<<<NPIX / 64, 256, 0, stream>>>(xT, Bo, b_off, pyA, pxA, mA);
  k_dcn_main<<<NPIX / 64, 256, 0, stream>>>(xT, Bw, pyA, pxA, mA, scl, bs2,
                                            out);
}
